// dnnCGM_89189290868766
// MI455X (gfx1250) — matrix operand order repaired and checked by reading, not compiled, not run on hardware
//
#include <hip/hip_runtime.h>
#include <hip/hip_bf16.h>

// Problem constants (from reference)
#define BB   256            // batch
#define CC   64             // z feature dim
#define HH   256            // hidden
#define NN   512            // N
#define NM1  511            // N-1
#define EE   261632         // N*(N-1)

typedef __attribute__((ext_vector_type(16))) __bf16        v16bf;
typedef __attribute__((ext_vector_type(8)))  float         v8f;
typedef __attribute__((ext_vector_type(4)))  unsigned int  v4u;
typedef __attribute__((ext_vector_type(4)))  int           v4i;
typedef __attribute__((ext_vector_type(8)))  int           v8i;

union Frag32B { uint4 u[2]; v16bf v; };

// TDM is mandatory this round: if __builtin_amdgcn_tensor_load_to_lds is not
// declared by this toolchain the compile fails (decisive probe). Both known
// gfx1250 toolchains (ROCm clang-22: 5 args, upstream clang-23: 6 args)
// declare it per the compile probes.
#if __clang_major__ >= 23
#  warning CDNA5_PROBE: TDM forced, 6-arg tensor_load_to_lds arity
#else
#  warning CDNA5_PROBE: TDM forced, 5-arg tensor_load_to_lds arity
#endif

static __device__ __forceinline__ unsigned short f32_to_bf16_rne(float f) {
    unsigned int u = __float_as_uint(f);
    u += 0x7FFFu + ((u >> 16) & 1u);      // round to nearest even
    return (unsigned short)(u >> 16);
}

// Pack two f32 -> two bf16 (round-half-up) in 3 VALU: 2x v_add + 1x v_perm.
// perm(S0=ub, S1=ua, 0x07060302): bytes {ub.b3, ub.b2, ua.b3, ua.b2}
//   -> low 16 = ua>>16, high 16 = ub>>16.
static __device__ __forceinline__ unsigned int pack2_bf16_rhu(float a, float b) {
    unsigned int ua = __float_as_uint(a) + 0x8000u;
    unsigned int ub = __float_as_uint(b) + 0x8000u;
    return __builtin_amdgcn_perm(ub, ua, 0x07060302u);
}

// ---------------------------------------------------------------------------
// Kernel 1: h2 = relu(relu(z@W1+b1)@W2+b2), stored as bf16 row-major [256][256]
// Tiny (~42 MFLOP) — plain f32 VALU is fine. One block owns 16 batch rows.
// ---------------------------------------------------------------------------
__global__ void mlp_kernel(const float* __restrict__ z,
                           const float* __restrict__ W1, const float* __restrict__ b1,
                           const float* __restrict__ W2, const float* __restrict__ b2,
                           unsigned short* __restrict__ h2b) {
    __shared__ float h1[16][HH + 1];
    const int t = threadIdx.x;           // 256 threads: one output column each
    const int rowBase = blockIdx.x * 16;

    for (int r = 0; r < 16; ++r) {
        float a = b1[t];
        #pragma unroll 8
        for (int k = 0; k < CC; ++k)
            a = fmaf(z[(rowBase + r) * CC + k], W1[k * HH + t], a);
        h1[r][t] = a > 0.f ? a : 0.f;
    }
    __syncthreads();
    for (int r = 0; r < 16; ++r) {
        float a = b2[t];
        #pragma unroll 8
        for (int k = 0; k < HH; ++k)
            a = fmaf(h1[r][k], W2[k * HH + t], a);
        a = a > 0.f ? a : 0.f;
        h2b[(size_t)(rowBase + r) * HH + t] = f32_to_bf16_rne(a);
    }
}

// ---------------------------------------------------------------------------
// Kernel 2 (fused): graph = h2@W3 + b3  AND  x_pred[b,i] = sum_j x[b,idx]*beta
// Grid: (512, 4). Block (i, ry) owns rows [ry*64, ry*64+64) x all 511 columns
// of target i (padded to 512, col 511 masked), so x_pred[b,i] reduces entirely
// in-block: no atomics, no 268MB graph re-read. 8 waves; wave w -> rowTile w%4,
// colTiles 4*(w/4)+{0..3}; 4x v8f accumulators per wave.
// A operand staged via TDM (tensor_load_to_lds with LDS row padding).
// ---------------------------------------------------------------------------
__global__ void __launch_bounds__(256)
graph_xpred_kernel(const unsigned short* __restrict__ h2b,   // [256][256] bf16
                   const float* __restrict__ W3,             // [256][E] f32
                   const float* __restrict__ b3,             // [E]
                   const float* __restrict__ x,              // [256][512]
                   float* __restrict__ xpred,                // [256][512]
                   float* __restrict__ graph) {              // [256][E]
    // LDS: A = h2 rows, bf16, row stride 264 shorts (512B data + 16B pad ->
    //      conflict-free b128 fragment reads). Bt = W3 tile transposed
    //      [col][k], stride 40 shorts (80B) so 16 lanes hit distinct bank sets.
    __shared__ unsigned short ldsA[64 * 264];    // 33,792 B
    __shared__ unsigned short ldsB[128 * 40];    // 10,240 B
    __shared__ float          ldsXP[2][64];

    const int tid  = threadIdx.x;
    const int lane = tid & 31;
    const int wave = tid >> 5;
    const int i       = blockIdx.x;           // target index, 0..511
    const int rowBase = blockIdx.y * 64;      // batch rows

    const int rowTile  = wave & 3;            // 0..3  -> rows rBase2..rBase2+15
    const int colGroup = wave >> 2;           // 0..1  -> colTiles 4*cg..4*cg+3
    const int rBase2   = rowTile * 16;
    const int rhalf    = (lane >> 4) << 3;    // 0 or 8 (C/D layout: M = v + rhalf)

    // ---- stage A via Tensor Data Mover: 2D tile 64 rows x 256 bf16 cols,
    //      LDS padding 16B per 512B row (matches ldsA stride 264 shorts).
    //      D# per ISA 8.3/8.4. TDM ignores EXEC; wave 0 issues, waits
    //      TENSORcnt, then the first workgroup barrier covers other waves.
    if (wave == 0) {
        unsigned long long ga =
            (unsigned long long)(uintptr_t)(h2b + (size_t)rowBase * HH);
        v4u g0 = { 1u,                                    // count=1, user D#
                   (unsigned int)(uintptr_t)&ldsA[0],     // lds_addr (bytes)
                   (unsigned int)(ga & 0xFFFFFFFFu),      // global_addr lo
                   (unsigned int)((ga >> 32) & 0x01FFFFFFu) | (2u << 30) }; // type=2
        v8i g1 = { (int)((1u << 16)      // data_size = 1 -> 2-byte elements
                       | (1u << 20)      // pad_enable (LDS dest padding)
                       | (6u << 22)      // pad_interval = 128 DWORDs (512B row)
                       | (3u << 25)),    // pad_amount   = 4 DWORDs   (16B)
                   (int)(256u << 16),    // tensor_dim0 = 256 (bits 63:48)
                   (int)(256u << 16),    // dim0_hi=0 | tensor_dim1 = 256
                   (int)(256u << 16),    // dim1_hi=0 | tile_dim0   = 256
                   (int)(64u),           // tile_dim1 = 64 | tile_dim2 = 0
                   (int)(256u),          // tensor_dim0_stride = 256 elements
                   0, 0 };               // dim1_stride unused for 2D tile
        v4i gz = { 0, 0, 0, 0 };
#if __clang_major__ >= 23
        v8i gz8 = { 0, 0, 0, 0, 0, 0, 0, 0 };
        __builtin_amdgcn_tensor_load_to_lds(g0, g1, gz, gz, gz8, 0);
#else
        __builtin_amdgcn_tensor_load_to_lds(g0, g1, gz, gz, 0);
#endif
    }
    __builtin_amdgcn_s_wait_tensorcnt(0);   // no-op for non-issuing waves

    float xp[8];
    #pragma unroll
    for (int v = 0; v < 8; ++v) xp[v] = 0.f;

    for (int cc = 0; cc < 4; ++cc) {          // 4 column chunks of 128
        const size_t ebase = (size_t)i * NM1 + cc * 128;
        v8f acc[4];
        #pragma unroll
        for (int ct = 0; ct < 4; ++ct)
            acc[ct] = (v8f){0.f,0.f,0.f,0.f,0.f,0.f,0.f,0.f};

        for (int ks = 0; ks < 8; ++ks) {      // K chunks of 32
            __syncthreads();                  // protect ldsB (and initial ldsA)

            // ---- stage B: W3[ks*32..+32)[128 cols] f32 -> bf16, LDS [col][k]
            #pragma unroll
            for (int rep = 0; rep < 2; ++rep) {
                int task = tid + rep * 256;   // 512 tasks: 16 k-pairs x 32 segs
                int kp  = task >> 5;          // k-pair 0..15
                int seg = task & 31;
                int klo = kp * 2;             // local k (even)
                size_t row0 = (size_t)(ks * 32 + klo) * EE + ebase;
                size_t row1 = row0 + EE;
                #pragma unroll
                for (int jj = 0; jj < 4; ++jj) {
                    int c = seg + jj * 32;    // lanes contiguous -> coalesced
                    bool valid = (cc * 128 + c) < NM1;
                    float f0 = valid ? W3[row0 + c] : 0.f;
                    float f1 = valid ? W3[row1 + c] : 0.f;
                    // 3 VALU per pair: 2x add (round-half-up) + 1x v_perm pack
                    *(unsigned int*)&ldsB[c * 40 + klo] = pack2_bf16_rhu(f0, f1);
                }
            }
            __syncthreads();

            // ---- A fragment (16x32 bf16, ISA 7.12.2): per half-wave the K
            // runs are {base..base+8} and {base+16..base+24} (base += 8 for
            // lanes 16-31) -> two b128 LDS loads.
            Frag32B af;
            {
                int r    = rBase2 + (lane & 15);
                int koff = ks * 32 + ((lane >> 4) << 3);
                af.u[0] = *(const uint4*)&ldsA[r * 264 + koff];
                af.u[1] = *(const uint4*)&ldsA[r * 264 + koff + 16];
            }

            // ---- all 4 B fragments first (graduated dscnt waits), then
            // 4 back-to-back WMMAs reusing the A fragment.
            Frag32B bfr[4];
            #pragma unroll
            for (int ct = 0; ct < 4; ++ct) {
                int c   = (colGroup * 4 + ct) * 16 + (lane & 15);
                int kfb = (lane >> 4) << 3;   // 0 or 8
                bfr[ct].u[0] = *(const uint4*)&ldsB[c * 40 + kfb];
                bfr[ct].u[1] = *(const uint4*)&ldsB[c * 40 + kfb + 16];
            }
            #pragma unroll
            for (int ct = 0; ct < 4; ++ct) {
                acc[ct] = __builtin_amdgcn_wmma_f32_16x16x32_bf16(
                              false, af.v, false, bfr[ct].v,
                              (short)0, acc[ct], false, false);
            }
        }

        // ---- epilogue for this column chunk: bias, store graph, x_pred part
        #pragma unroll
        for (int ct = 0; ct < 4; ++ct) {
            int ctg = colGroup * 4 + ct;
            int j = cc * 128 + ctg * 16 + (lane & 15);   // local col = j index
            if (j < NM1) {
                size_t e = (size_t)i * NM1 + j;
                float bias = b3[e];
                int jidx = j + (j >= i);                  // self-removed gather idx
                #pragma unroll
                for (int v = 0; v < 8; ++v) {
                    int brow = rowBase + rBase2 + rhalf + v;
                    float g = acc[ct][v] + bias;
                    graph[(size_t)brow * EE + e] = g;
                    xp[v] = fmaf(g, x[brow * NN + jidx], xp[v]);
                }
            }
        }
    }

    // ---- reduce x_pred across the 16 lanes sharing each row (wave32) ----
    #pragma unroll
    for (int v = 0; v < 8; ++v) {
        float s = xp[v];
        s += __shfl_xor(s, 1, 16);
        s += __shfl_xor(s, 2, 16);
        s += __shfl_xor(s, 4, 16);
        s += __shfl_xor(s, 8, 16);
        xp[v] = s;
    }
    if ((lane & 15) == 0) {
        int base = rBase2 + rhalf;
        #pragma unroll
        for (int v = 0; v < 8; ++v) {
            if (colGroup == 0) ldsXP[0][base + v] = xp[v];
            else               ldsXP[1][base + v] = xp[v];
        }
    }
    __syncthreads();
    if (tid < 64) {
        xpred[(size_t)(rowBase + tid) * NN + i] = ldsXP[0][tid] + ldsXP[1][tid];
    }
}

// ---------------------------------------------------------------------------
extern "C" void kernel_launch(void* const* d_in, const int* in_sizes, int n_in,
                              void* d_out, int out_size, void* d_ws, size_t ws_size,
                              hipStream_t stream) {
    const float* x  = (const float*)d_in[0];   // (256,512)
    const float* z  = (const float*)d_in[1];   // (256,64)
    const float* W1 = (const float*)d_in[2];   // (64,256)
    const float* b1 = (const float*)d_in[3];   // (256,)
    const float* W2 = (const float*)d_in[4];   // (256,256)
    const float* b2 = (const float*)d_in[5];   // (256,)
    const float* W3 = (const float*)d_in[6];   // (256,261632)
    const float* b3 = (const float*)d_in[7];   // (261632,)

    float* xpred = (float*)d_out;                    // first output: (256,512)
    float* graph = (float*)d_out + (size_t)BB * NN;  // second: (256,261632)

    unsigned short* h2b = (unsigned short*)d_ws;     // 256x256 bf16 = 128 KB

    mlp_kernel<<<16, 256, 0, stream>>>(z, W1, b1, W2, b2, h2b);

    dim3 grid(NN, BB / 64);                    // (512 targets, 4 row tiles)
    graph_xpred_kernel<<<grid, 256, 0, stream>>>(h2b, W3, b3, x, xpred, graph);
}
